// CapsuleLayer_28724741275833
// MI455X (gfx1250) — matrix operand order repaired and checked by reading, not compiled, not run on hardware
//
#include <hip/hip_runtime.h>
#include <cstdint>

// ---------------- problem constants ----------------
#define B_   64
#define NIN  512
#define DIN  1024
#define NCAP 16
#define DCAP 64
#define M_   (B_ * NIN)      // 32768
#define N_   (NCAP * DCAP)   // 1024
#define K_   DIN             // 1024

// ---------------- vector types ----------------
typedef __bf16 v16bf __attribute__((ext_vector_type(16)));
typedef __bf16 v8bf  __attribute__((ext_vector_type(8)));
typedef float  v8f   __attribute__((ext_vector_type(8)));

// ---------------- GEMM tiling ----------------
#define TM   128
#define TN   128
#define TK   32
#define ASTR 40   // padded LDS row stride (elements) -> 80B rows, bank-conflict free
#define BSTR 40

// CDNA5 async global->LDS path (ASYNCcnt), if the toolchain exposes it
#if defined(__AMDGCN__) && __has_builtin(__builtin_amdgcn_global_load_async_to_lds_b128)
#define ASYNC_LDS 1
// exact parameter types per hipcc diagnostic: int __vector(4) in AS1 / AS3
typedef int async_v4i __attribute__((vector_size(16)));
typedef __attribute__((address_space(1))) async_v4i* async_gp;
typedef __attribute__((address_space(3))) async_v4i* async_lp;
#else
#define ASYNC_LDS 0
#endif

// =====================================================================
// fp32 -> bf16 conversion of x (8 elements / thread)
// =====================================================================
__global__ __launch_bounds__(256)
void k_cvt_x(const float* __restrict__ x, __bf16* __restrict__ xb) {
    size_t idx  = (size_t)blockIdx.x * 256 + threadIdx.x;
    size_t base = idx * 8;
    float4 f0 = *(const float4*)(x + base);
    float4 f1 = *(const float4*)(x + base + 4);
    v8bf o;
    o[0] = (__bf16)f0.x; o[1] = (__bf16)f0.y; o[2] = (__bf16)f0.z; o[3] = (__bf16)f0.w;
    o[4] = (__bf16)f1.x; o[5] = (__bf16)f1.y; o[6] = (__bf16)f1.z; o[7] = (__bf16)f1.w;
    *(v8bf*)(xb + base) = o;
}

// =====================================================================
// fp32 W[k][n] -> bf16 Wt[n][k]  (transpose so B-frags are contiguous in K)
// =====================================================================
__global__ __launch_bounds__(256)
void k_cvt_wt(const float* __restrict__ W, __bf16* __restrict__ wt) {
    int idx = blockIdx.x * 256 + threadIdx.x;      // over K_*N_
    int n = idx & (N_ - 1);
    int k = idx >> 10;
    wt[(size_t)n * K_ + k] = (__bf16)W[idx];
}

// =====================================================================
// u_hat = x @ W  via v_wmma_f32_16x16x32_bf16, epilogue writes the
// permuted layout u_hat[b][cap][nin][dim] directly.
// block: 256 thr = 8 waves; block tile 128x128; wave tile 64x32 (4x2 frags)
// =====================================================================
__global__ __launch_bounds__(256)
void k_gemm_bf16_wmma(const __bf16* __restrict__ xb,
                      const __bf16* __restrict__ wt,
                      float* __restrict__ uhat) {
    __shared__ __bf16 As[2][TM * ASTR];
    __shared__ __bf16 Bs[2][TN * BSTR];

    const int tid  = threadIdx.x;
    const int lane = tid & 31;
    const int w    = tid >> 5;
    const int wm   = w & 1;        // 2 waves along M
    const int wn   = w >> 1;       // 4 waves along N
    const int m0   = blockIdx.y * TM;
    const int n0   = blockIdx.x * TN;
    const int g    = lane >> 4;    // half-wave group
    const int r    = lane & 15;

    v8f acc[4][2];
#pragma unroll
    for (int i = 0; i < 4; ++i)
#pragma unroll
        for (int j = 0; j < 2; ++j) acc[i][j] = (v8f){};

    // stage one (TM x TK) A tile and (TN x TK) B tile into LDS
    auto fill = [&](int buf, int kt) {
#pragma unroll
        for (int c = 0; c < 2; ++c) {
            int idx = c * 256 + tid;      // 512 chunks of 16B per tile
            int row = idx >> 2;           // 0..127
            int ch  = idx & 3;            // 4 x 16B = 32 elems per row
            const __bf16* pga = xb + (size_t)(m0 + row) * K_ + kt + ch * 8;
            const __bf16* pgb = wt + (size_t)(n0 + row) * K_ + kt + ch * 8;
            if (kt + 2 * TK < K_) {       // warm L2/L0 for kt+2
                __builtin_prefetch(pga + 2 * TK, 0, 1);
                __builtin_prefetch(pgb + 2 * TK, 0, 1);
            }
#if ASYNC_LDS
            // VGPR-bypassing async copy, tracked by ASYNCcnt.
            // AS1 global pointer is numerically identical to generic;
            // AS3 LDS offset is the low 32 bits of the generic pointer.
            __builtin_amdgcn_global_load_async_to_lds_b128(
                (async_gp)(uintptr_t)pga,
                (async_lp)(uint32_t)(uintptr_t)&As[buf][row * ASTR + ch * 8],
                0, 0);
            __builtin_amdgcn_global_load_async_to_lds_b128(
                (async_gp)(uintptr_t)pgb,
                (async_lp)(uint32_t)(uintptr_t)&Bs[buf][row * BSTR + ch * 8],
                0, 0);
#else
            *(uint4*)(&As[buf][row * ASTR + ch * 8]) = *(const uint4*)pga;
            *(uint4*)(&Bs[buf][row * BSTR + ch * 8]) = *(const uint4*)pgb;
#endif
        }
    };

    auto wait_fill = [&]() {
#if ASYNC_LDS
#if __has_builtin(__builtin_amdgcn_s_wait_asynccnt)
        __builtin_amdgcn_s_wait_asynccnt(0);
#else
        asm volatile("s_wait_asynccnt 0x0" ::: "memory");
#endif
#endif
    };

    fill(0, 0);
    wait_fill();
    __syncthreads();

    int buf = 0;
    for (int kt = 0; kt < K_ / TK; ++kt) {
        if (kt + 1 < K_ / TK) fill(buf ^ 1, (kt + 1) * TK);

        // A fragments: per ISA 16-bit A 16x32 layout:
        // lane group g=0: K 0..7 & 16..23 ; g=1: K 8..15 & 24..31 ; M = lane&15
        v16bf af[4];
#pragma unroll
        for (int fm = 0; fm < 4; ++fm) {
            const __bf16* p = &As[buf][(wm * 64 + fm * 16 + r) * ASTR + g * 8];
            union { v16bf v; v8bf h[2]; } u;
            u.h[0] = *(const v8bf*)p;
            u.h[1] = *(const v8bf*)(p + 16);
            af[fm] = u.v;
        }
        // B fragments: N = lane&15, K = g*8..g*8+7 & 16+g*8..16+g*8+7 (Wt is [n][k])
        v16bf bfr[2];
#pragma unroll
        for (int fn = 0; fn < 2; ++fn) {
            const __bf16* p = &Bs[buf][(wn * 32 + fn * 16 + r) * BSTR + g * 8];
            union { v16bf v; v8bf h[2]; } u;
            u.h[0] = *(const v8bf*)p;
            u.h[1] = *(const v8bf*)(p + 16);
            bfr[fn] = u.v;
        }
#pragma unroll
        for (int fm = 0; fm < 4; ++fm)
#pragma unroll
            for (int fn = 0; fn < 2; ++fn)
                acc[fm][fn] = __builtin_amdgcn_wmma_f32_16x16x32_bf16(
                    false, af[fm], false, bfr[fn], (short)0, acc[fm][fn],
                    false, false);

        wait_fill();          // our async fills for buf^1 are done
        __syncthreads();      // everyone's fills done; reads of buf done
        buf ^= 1;
    }

    // epilogue: C/D layout -> lanes 0-15: M=i, lanes 16-31: M=8+i; N=lane&15
    // store into u_hat[b][cap][nin][dim]  (m=b*512+nin, n=cap*64+dim)
#pragma unroll
    for (int fm = 0; fm < 4; ++fm) {
#pragma unroll
        for (int fn = 0; fn < 2; ++fn) {
            int mb  = m0 + wm * 64 + fm * 16 + g * 8;
            int nn  = n0 + wn * 32 + fn * 16 + r;
            int cap = nn >> 6;
            int dim = nn & (DCAP - 1);
#pragma unroll
            for (int i = 0; i < 8; ++i) {
                int m    = mb + i;
                int bidx = m >> 9;
                int nin  = m & (NIN - 1);
                uhat[(((size_t)bidx * NCAP + cap) * NIN + nin) * DCAP + dim] =
                    acc[fm][fn][i];
            }
        }
    }
}

// =====================================================================
// c = softmax over the 16 capsules (axis=1). one thread per (b, j).
// =====================================================================
__global__ __launch_bounds__(256)
void k_softmax_c(const float* __restrict__ bl, float* __restrict__ cc) {
    int idx = blockIdx.x * 256 + threadIdx.x;  // B_*NIN
    int b = idx >> 9;
    int j = idx & (NIN - 1);
    const float* p = bl + (size_t)b * NCAP * NIN + j;
    float v[NCAP], mx = -3.0e38f;
#pragma unroll
    for (int i = 0; i < NCAP; ++i) { v[i] = p[(size_t)i * NIN]; mx = fmaxf(mx, v[i]); }
    float s = 0.f;
#pragma unroll
    for (int i = 0; i < NCAP; ++i) { v[i] = __expf(v[i] - mx); s += v[i]; }
    float inv = 1.f / s;
    float* q = cc + (size_t)b * NCAP * NIN + j;
#pragma unroll
    for (int i = 0; i < NCAP; ++i) q[(size_t)i * NIN] = v[i] * inv;
}

// =====================================================================
// outputs[b,i,k] = squash( sum_j c[b,i,j] * u_hat[b,i,j,k] )
// block = (b,i); 256 thr = 64 k-lanes x 4 j-groups.
// =====================================================================
__global__ __launch_bounds__(256)
void k_outputs(const float* __restrict__ uhat, const float* __restrict__ cc,
               float* __restrict__ out, int uniformC) {
    const int bi  = blockIdx.x;           // b*16 + i
    const int tid = threadIdx.x;
    const int k   = tid & 63;
    const int jg  = tid >> 6;
    const float* up = uhat + (size_t)bi * NIN * DCAP;
    const float* cp = cc + (size_t)bi * NIN;

    float s = 0.f;
    for (int jj = 0; jj < NIN / 4; ++jj) {
        int j   = jj * 4 + jg;
        float c = uniformC ? 0.0625f : cp[j];
        s = fmaf(c, up[(size_t)j * DCAP + k], s);
    }
    __shared__ float red[256];
    __shared__ float nrmS;
    red[tid] = s;
    __syncthreads();
    float sk = 0.f;
    if (tid < 64) {
        sk = red[tid] + red[tid + 64] + red[tid + 128] + red[tid + 192];
        red[tid] = sk;
    }
    __syncthreads();
    if (tid < 32) {
        float t = red[tid] * red[tid] + red[tid + 32] * red[tid + 32];
        t += __shfl_xor(t, 16, 32);
        t += __shfl_xor(t, 8, 32);
        t += __shfl_xor(t, 4, 32);
        t += __shfl_xor(t, 2, 32);
        t += __shfl_xor(t, 1, 32);
        if (tid == 0) nrmS = t;
    }
    __syncthreads();
    if (tid < 64)
        out[(size_t)bi * DCAP + tid] = red[tid] * rsqrtf(nrmS + 1e-7f);
}

// =====================================================================
// b[b,i,j] = sum_k outputs[b,i,k] * u_hat[b,i,j,k]
// block = (b,i); each wave: 8 j x 4 k-groups of 16; shfl_xor reduce.
// =====================================================================
__global__ __launch_bounds__(256)
void k_update_b(const float* __restrict__ uhat, const float* __restrict__ out,
                float* __restrict__ bl) {
    const int bi   = blockIdx.x;
    const int tid  = threadIdx.x;
    const int lane = tid & 31;
    const int w    = tid >> 5;
    const int jj   = lane >> 2;   // 0..7
    const int kg   = lane & 3;    // 0..3
    const float* up = uhat + (size_t)bi * NIN * DCAP;
    const float* op = out + (size_t)bi * DCAP + kg * 16;

    float4 o0 = *(const float4*)(op + 0);
    float4 o1 = *(const float4*)(op + 4);
    float4 o2 = *(const float4*)(op + 8);
    float4 o3 = *(const float4*)(op + 12);

    for (int t = 0; t < NIN / 64; ++t) {
        int j = t * 64 + w * 8 + jj;
        const float* rp = up + (size_t)j * DCAP + kg * 16;
        float4 a0 = *(const float4*)(rp + 0);
        float4 a1 = *(const float4*)(rp + 4);
        float4 a2 = *(const float4*)(rp + 8);
        float4 a3 = *(const float4*)(rp + 12);
        float d = a0.x * o0.x + a0.y * o0.y + a0.z * o0.z + a0.w * o0.w
                + a1.x * o1.x + a1.y * o1.y + a1.z * o1.z + a1.w * o1.w
                + a2.x * o2.x + a2.y * o2.y + a2.z * o2.z + a2.w * o2.w
                + a3.x * o3.x + a3.y * o3.y + a3.z * o3.z + a3.w * o3.w;
        d += __shfl_xor(d, 1, 32);
        d += __shfl_xor(d, 2, 32);
        if (kg == 0) bl[(size_t)bi * NIN + j] = d;
    }
}

// =====================================================================
extern "C" void kernel_launch(void* const* d_in, const int* in_sizes, int n_in,
                              void* d_out, int out_size, void* d_ws, size_t ws_size,
                              hipStream_t stream) {
    const float* x = (const float*)d_in[0];
    const float* W = (const float*)d_in[1];
    float* out = (float*)d_out;
    char* ws = (char*)d_ws;

    const size_t XB = (size_t)M_ * K_ * 2;            // 64 MB  x in bf16
    const size_t WT = (size_t)N_ * K_ * 2;            // 2 MB   W^T in bf16
    const size_t UH = (size_t)M_ * N_ * 4;            // 134 MB u_hat fp32
    const size_t BL = (size_t)B_ * NCAP * NIN * 4;    // 2 MB

    __bf16* xb   = (__bf16*)(ws);
    __bf16* wt   = (__bf16*)(ws + XB);
    float*  uhat = (float*)(ws + XB + WT);
    float*  bl   = (float*)(ws + XB + WT + UH);
    float*  cc   = (float*)(ws + XB + WT + UH + BL);

    // 1) precision staging
    k_cvt_x <<<(M_ * K_ / 8) / 256, 256, 0, stream>>>(x, xb);
    k_cvt_wt<<<(K_ * N_) / 256,    256, 0, stream>>>(W, wt);

    // 2) u_hat GEMM (WMMA bf16, f32 accumulate), permuted epilogue
    dim3 gg(N_ / TN, M_ / TM);
    k_gemm_bf16_wmma<<<gg, 256, 0, stream>>>(xb, wt, uhat);

    // 3) dynamic routing (3 iterations; iter0 softmax(0) == 1/16 exactly)
    k_outputs <<<B_ * NCAP, 256, 0, stream>>>(uhat, cc, out, 1);
    k_update_b<<<B_ * NCAP, 256, 0, stream>>>(uhat, out, bl);

    k_softmax_c<<<(B_ * NIN) / 256, 256, 0, stream>>>(bl, cc);
    k_outputs <<<B_ * NCAP, 256, 0, stream>>>(uhat, cc, out, 0);
    k_update_b<<<B_ * NCAP, 256, 0, stream>>>(uhat, out, bl);

    k_softmax_c<<<(B_ * NIN) / 256, 256, 0, stream>>>(bl, cc);
    k_outputs <<<B_ * NCAP, 256, 0, stream>>>(uhat, cc, out, 0);

    (void)in_sizes; (void)n_in; (void)out_size; (void)ws_size;
}
